// TypeLayer_36524401885446
// MI455X (gfx1250) — matrix operand order repaired and checked by reading, not compiled, not run on hardware
//
#include <hip/hip_runtime.h>

typedef __attribute__((ext_vector_type(16))) _Float16 v16h;
typedef __attribute__((ext_vector_type(8)))  _Float16 v8h;
typedef __attribute__((ext_vector_type(8)))  float    v8f;

#define HDIM 128

// ---------------------------------------------------------------------------
// Kernel 1: convert rel_features and W to f16 (for WMMA) and zero the output
// accumulator (d_out is poisoned by the harness).
// ---------------------------------------------------------------------------
__global__ void prep_kernel(const float* __restrict__ relf,
                            const float* __restrict__ Wf,
                            _Float16* __restrict__ Ah,
                            _Float16* __restrict__ Wh,
                            float* __restrict__ out,
                            int nA, int nW, int nOut) {
    int tid = blockIdx.x * blockDim.x + threadIdx.x;
    if (tid < nA)   Ah[tid] = (_Float16)relf[tid];
    if (tid < nW)   Wh[tid] = (_Float16)Wf[tid];
    if (tid < nOut) out[tid] = 0.0f;
}

// ---------------------------------------------------------------------------
// Kernel 2: rel_val[r, j] = sum_k rel_features[r,k] * W[j,k] + b[j]
// One wave (32 threads) computes one 16x16 tile of rel_val via
// v_wmma_f32_16x16x32_f16, K-loop of 128/32 = 4 iterations.
//
// A layout (16-bit A 16x32, contiguous k order): lane L<16 -> row M=L, VGPR0-3 hold
// K=khalf*16..khalf*16+7, VGPR4-7 hold K=khalf*16+8..khalf*16+15, khalf = lane>>4.
// B layout (16-bit B 32x16, contiguous k, same order as A): lanes 0-15 hold K=0..15
// (2 per VGPR), lanes 16-31 hold K=16..31; column N = lane&15.
// Since B[k][n] = W[n][k], lane n loads 16 contiguous halves of W row n.
// ---------------------------------------------------------------------------
__global__ void relval_gemm_wmma(const _Float16* __restrict__ Ah,
                                 const _Float16* __restrict__ Wh,
                                 const float* __restrict__ bias,
                                 float* __restrict__ relval) {
    const int lane  = threadIdx.x & 31;
    const int tn    = blockIdx.x & 7;        // HDIM/16 = 8 column tiles
    const int tm    = blockIdx.x >> 3;       // row tile over relations
    const int r15   = lane & 15;
    const int khalf = lane >> 4;

    const int arow = tm * 16 + r15;          // relation row
    const int brow = tn * 16 + r15;          // W row == output column

    union { v16h v; v8h h[2]; } a;
    v8f c = {};

    const _Float16* aptr = Ah + (size_t)arow * HDIM + khalf * 16;
    const _Float16* bptr = Wh + (size_t)brow * HDIM + khalf * 16;

    #pragma unroll
    for (int kc = 0; kc < HDIM / 32; ++kc) {
        a.h[0]   = *(const v8h*)(aptr + kc * 32);
        a.h[1]   = *(const v8h*)(aptr + kc * 32 + 8);
        v16h bm  = *(const v16h*)(bptr + kc * 32);
        c = __builtin_amdgcn_wmma_f32_16x16x32_f16(
                /*neg_a=*/false, a.v, /*neg_b=*/false, bm,
                /*c_mod=*/(short)0, c, /*reuse_a=*/false, /*reuse_b=*/false);
    }

    const float bn    = bias[brow];
    const int   mbase = tm * 16 + 8 * khalf;
    #pragma unroll
    for (int v = 0; v < 8; ++v) {
        relval[(size_t)(mbase + v) * HDIM + brow] = c[v] + bn;
    }
}

// ---------------------------------------------------------------------------
// Kernel 3: scatter-add. One wave per fact iteration; each lane covers 4
// columns (lane, lane+32, lane+64, lane+96) -> fully coalesced 128B atomic
// bursts. rel_val (3 MB) and out (8 MB) are L2-resident.
// ---------------------------------------------------------------------------
__device__ __forceinline__ void gatomic_add_f32(float* addr, float v) {
    asm volatile("global_atomic_add_f32 %0, %1, off scope:SCOPE_DEV"
                 :: "v"(addr), "v"(v) : "memory");
}

__global__ void scatter_kernel(const float* __restrict__ relval,
                               const int* __restrict__ rels,
                               const int* __restrict__ heads,
                               const int* __restrict__ tails,
                               const float* __restrict__ wvals,
                               float* __restrict__ out,
                               int nfact) {
    const int lane  = threadIdx.x & 31;
    const int wave  = (blockIdx.x * blockDim.x + threadIdx.x) >> 5;
    const int nwave = (gridDim.x * blockDim.x) >> 5;
    for (int f = wave; f < nfact; f += nwave) {
        const int   r = rels[f];
        const int   h = heads[f];
        const int   t = tails[f];
        const float w = wvals[f];
        const float* src = relval + (size_t)r * HDIM;
        float* oh = out + (size_t)h * HDIM;
        float* ot = out + (size_t)t * HDIM;
        #pragma unroll
        for (int i = 0; i < 4; ++i) {
            const int col = lane + 32 * i;
            const float v = src[col] * w;
            gatomic_add_f32(ot + col, v);
            gatomic_add_f32(oh + col, v);
        }
    }
}

// ---------------------------------------------------------------------------
// Kernel 4: relu (runs after all atomics via kernel-boundary ordering).
// ---------------------------------------------------------------------------
__global__ void relu_kernel(float* __restrict__ out, int n) {
    int tid = blockIdx.x * blockDim.x + threadIdx.x;
    if (tid < n) out[tid] = fmaxf(out[tid], 0.0f);
}

// ---------------------------------------------------------------------------
extern "C" void kernel_launch(void* const* d_in, const int* in_sizes, int n_in,
                              void* d_out, int out_size, void* d_ws, size_t ws_size,
                              hipStream_t stream) {
    // setup_inputs order:
    // 0: local_entity (unused, shapes are fixed)
    // 1: batch_heads, 2: batch_rels, 3: batch_tails, 4: weight_rel_list
    // 5: rel_features [NREL,128], 6: W [128,128], 7: b [128]
    const int*   heads = (const int*)d_in[1];
    const int*   rels  = (const int*)d_in[2];
    const int*   tails = (const int*)d_in[3];
    const float* wvals = (const float*)d_in[4];
    const float* relf  = (const float*)d_in[5];
    const float* Wf    = (const float*)d_in[6];
    const float* bias  = (const float*)d_in[7];
    float* out = (float*)d_out;

    const int nfact = in_sizes[1];
    const int nrel  = in_sizes[5] / HDIM;   // 6000
    const int nA    = nrel * HDIM;          // 768000 halves
    const int nW    = HDIM * HDIM;          // 16384 halves
    const int nOut  = out_size;             // 2,048,000 floats

    // Workspace layout (all 32B aligned):
    char* ws = (char*)d_ws;
    _Float16* Ah     = (_Float16*)ws;                               // nA * 2 B
    _Float16* Wh     = (_Float16*)(ws + (size_t)nA * 2);            // nW * 2 B
    float*    relval = (float*)(ws + (size_t)nA * 2 + (size_t)nW * 2); // nA * 4 B

    prep_kernel<<<(nOut + 255) / 256, 256, 0, stream>>>(
        relf, Wf, Ah, Wh, out, nA, nW, nOut);

    const int tiles = (nrel / 16) * (HDIM / 16);   // 375 * 8 = 3000 waves
    relval_gemm_wmma<<<tiles, 32, 0, stream>>>(Ah, Wh, bias, relval);

    scatter_kernel<<<1024, 256, 0, stream>>>(
        relval, rels, heads, tails, wvals, out, nfact);

    relu_kernel<<<(nOut + 255) / 256, 256, 0, stream>>>(out, nOut);
}
